// SimSVDCrossVolume_42219528520134
// MI455X (gfx1250) — matrix operand order repaired and checked by reading, not compiled, not run on hardware
//
#include <hip/hip_runtime.h>
#include <hip/hip_bf16.h>
#include <math.h>

typedef __attribute__((ext_vector_type(16))) _Float16 v16h;
typedef __attribute__((ext_vector_type(8)))  float    v8f;

#define NPTS 256
#define MPIX 1024
#define KNN  16

// ---------------- workspace layout (float offsets) ----------------
constexpr size_t OXYZ = 0;                       // 256*3
constexpr size_t OSN  = OXYZ + 256*3;            // 256*64  normalized warped_points
constexpr size_t ODN  = OSN  + 256*64;           // 1024*64 normalized rf
constexpr size_t ORFI = ODN  + 1024*64;          // 1024*3  rfi transposed (m-major)
constexpr size_t OS1  = ORFI + 1024*3;           // 256*1024 sim1[n][m]
constexpr size_t ORM1 = OS1  + 256*1024;         // 256
constexpr size_t OCM1 = ORM1 + 256;              // 1024
constexpr size_t OPCF = OCM1 + 1024;             // 256*64  pc_feats
constexpr size_t OPCN = OPCF + 256*64;           // 256*64
constexpr size_t OIMF = OPCN + 256*64;           // 1024*64 img_feats
constexpr size_t OIMN = OIMF + 1024*64;          // 1024*64
constexpr size_t OS2  = OIMN + 1024*64;          // 256*1024 sim2[n][m]
constexpr size_t ORM2 = OS2  + 256*1024;         // 256
constexpr size_t OCM2 = ORM2 + 256;              // 1024
constexpr size_t OAN  = OCM2 + 1024;             // 256*128  An (incl. b1_0)
constexpr size_t OBM  = OAN  + 256*128;          // 1024*128 Bm
constexpr size_t OATT = OBM  + 1024*128;         // 256*64
constexpr size_t OCOR = OATT + 256*64;           // 256*3
constexpr size_t OWIN = OCOR + 256*3;            // 256

// ---------------- prep: xyz, normalized src/dst rows, rfi transpose ----------------
__global__ void k_prep(const float* wxyz, const float* wp, const float* rf3,
                       const float* rf3i, const float* lz,
                       float* xyz, float* snorm, float* dnorm, float* rfit) {
  __shared__ float red[64];
  int b = blockIdx.x, t = threadIdx.x;
  if (b < NPTS) {
    int n = b;
    if (t < 3) xyz[n*3+t] = wxyz[n*3+t] * lz[n];
    float v = wp[n*64+t];
    red[t] = v*v; __syncthreads();
    for (int s=32;s>0;s>>=1){ if(t<s) red[t]+=red[t+s]; __syncthreads(); }
    snorm[n*64+t] = v / fmaxf(sqrtf(red[0]), 1e-12f);
  } else {
    int m = b - NPTS;
    if (t < 3) rfit[m*3+t] = rf3i[t*MPIX+m];
    float v = rf3[t*MPIX+m];
    red[t] = v*v; __syncthreads();
    for (int s=32;s>0;s>>=1){ if(t<s) red[t]+=red[t+s]; __syncthreads(); }
    dnorm[m*64+t] = v / fmaxf(sqrtf(red[0]), 1e-12f);
  }
}

// ---------------- sim[n][m] = dot(a[n], b[m]) ----------------
__global__ void k_sim(const float* a, const float* bm, float* sim) {
  __shared__ float arow[64];
  int n = blockIdx.x, t = threadIdx.x;
  if (t < 64) arow[t] = a[n*64+t];
  __syncthreads();
  for (int m = t; m < MPIX; m += 256) {
    const float* br = bm + (size_t)m*64;
    float s = 0.f;
    for (int c = 0; c < 64; ++c) s += arow[c]*br[c];
    sim[(size_t)n*MPIX + m] = s;
  }
}

// ---------------- row max (over m) and col max (over n) ----------------
__global__ void k_maxes(const float* sim, float* rmax, float* cmax) {
  __shared__ float red[256];
  int b = blockIdx.x, t = threadIdx.x;
  if (b < NPTS) {
    float mx = -1e30f;
    for (int m = t; m < MPIX; m += 256) mx = fmaxf(mx, sim[(size_t)b*MPIX+m]);
    red[t]=mx; __syncthreads();
    for (int s=128;s>0;s>>=1){ if(t<s) red[t]=fmaxf(red[t],red[t+s]); __syncthreads(); }
    if (t==0) rmax[b]=red[0];
  } else {
    int m = b - NPTS;
    float mx = (t < NPTS) ? sim[(size_t)t*MPIX+m] : -1e30f;
    red[t]=mx; __syncthreads();
    for (int s=128;s>0;s>>=1){ if(t<s) red[t]=fmaxf(red[t],red[t+s]); __syncthreads(); }
    if (t==0) cmax[m]=red[0];
  }
}

// ---------------- KNN + MLP2 + softmax-weighted pc_feats ----------------
__global__ void k_knn(const float* xyz, const float* wp,
                      const float* w20,const float* b20,const float* w21,const float* b21,
                      const float* w22,const float* b22, float* pcf) {
  __shared__ float d2s[256];
  __shared__ float rv[64]; __shared__ int ri[64];
  __shared__ int idxs[KNN];
  __shared__ float nin[68], h1[64], h2[64], red[64], wmaxs[KNN];
  int n = blockIdx.x, t = threadIdx.x;
  float px=xyz[n*3], py=xyz[n*3+1], pz=xyz[n*3+2];
  for (int q=t;q<256;q+=64){
    float dx=xyz[q*3]-px, dy=xyz[q*3+1]-py, dz=xyz[q*3+2]-pz;
    d2s[q]=dx*dx+dy*dy+dz*dz;
  }
  __syncthreads();
  for (int k=0;k<KNN;++k){
    float bv=1e30f; int bi=0;
    for (int q=t;q<256;q+=64){
      float v=d2s[q];
      if (v<bv || (v==bv && q<bi)) { bv=v; bi=q; }
    }
    rv[t]=bv; ri[t]=bi; __syncthreads();
    for (int s=32;s>0;s>>=1){
      if (t<s){
        float ov=rv[t+s]; int oi=ri[t+s];
        if (ov<rv[t] || (ov==rv[t] && oi<ri[t])){ rv[t]=ov; ri[t]=oi; }
      }
      __syncthreads();
    }
    if (t==0){ idxs[k]=ri[0]; d2s[ri[0]]=1e30f; }
    __syncthreads();
  }
  for (int k=0;k<KNN;++k){
    int j = idxs[k];
    nin[t] = wp[j*64+t];
    if (t==0){
      float dx=xyz[j*3]-px, dy=xyz[j*3+1]-py, dz=xyz[j*3+2]-pz;
      nin[64]=dx; nin[65]=dy; nin[66]=dz; nin[67]=sqrtf(dx*dx+dy*dy+dz*dz);
    }
    __syncthreads();
    float a=b20[t];
    for (int c=0;c<68;++c) a += nin[c]*w20[c*64+t];
    h1[t]=fmaxf(a,0.f); __syncthreads();
    a=b21[t];
    for (int c=0;c<64;++c) a += h1[c]*w21[c*64+t];
    h2[t]=fmaxf(a,0.f); __syncthreads();
    a=b22[t];
    for (int c=0;c<64;++c) a += h2[c]*w22[c*64+t];
    red[t]=fmaxf(a,0.f); __syncthreads();
    for (int s=32;s>0;s>>=1){ if(t<s) red[t]=fmaxf(red[t],red[t+s]); __syncthreads(); }
    if (t==0) wmaxs[k]=red[0];
    __syncthreads();
  }
  float mx=-1e30f;
  for (int k=0;k<KNN;++k) mx=fmaxf(mx,wmaxs[k]);
  float den=0.f, sm[KNN];
  for (int k=0;k<KNN;++k){ sm[k]=__expf(wmaxs[k]-mx); den+=sm[k]; }
  float acc=0.f;
  for (int k=0;k<KNN;++k) acc += (sm[k]/den)*wp[idxs[k]*64+t];
  pcf[n*64+t]=acc;
}

// ---------------- unfold + MLP3 + softmax-weighted img_feats ----------------
__global__ void k_img(const float* rf3, const float* rf3i,
                      const float* w30,const float* b30,const float* w31,const float* b31,
                      const float* w32,const float* b32, float* imf){
  __shared__ float nin[67], h1[64], h2[64], red[64], iwv[9];
  int m = blockIdx.x, t = threadIdx.x;
  int h = m >> 5, w = m & 31;
  for (int p=0;p<9;++p){
    int hh = h + p/3 - 1, ww = w + p%3 - 1;
    bool ok = (hh>=0 && hh<32 && ww>=0 && ww<32);
    int mi = hh*32+ww;
    if (t<3) nin[t] = ok ? rf3i[t*MPIX+mi] : 0.f;
    nin[3+t] = ok ? rf3[t*MPIX+mi] : 0.f;
    __syncthreads();
    float a=b30[t];
    for (int c=0;c<67;++c) a += nin[c]*w30[c*64+t];
    h1[t]=fmaxf(a,0.f); __syncthreads();
    a=b31[t];
    for (int c=0;c<64;++c) a += h1[c]*w31[c*64+t];
    h2[t]=fmaxf(a,0.f); __syncthreads();
    a=b32[t];
    for (int c=0;c<64;++c) a += h2[c]*w32[c*64+t];
    red[t]=fmaxf(a,0.f); __syncthreads();
    for (int s=32;s>0;s>>=1){ if(t<s) red[t]=fmaxf(red[t],red[t+s]); __syncthreads(); }
    if (t==0) iwv[p]=red[0];
    __syncthreads();
  }
  float mx=-1e30f;
  for (int p=0;p<9;++p) mx=fmaxf(mx,iwv[p]);
  float den=0.f, sm[9];
  for (int p=0;p<9;++p){ sm[p]=__expf(iwv[p]-mx); den+=sm[p]; }
  float acc=0.f;
  for (int p=0;p<9;++p){
    int hh=h+p/3-1, ww=w+p%3-1;
    bool ok=(hh>=0&&hh<32&&ww>=0&&ww<32);
    float v = ok ? rf3[t*MPIX + hh*32+ww] : 0.f;
    acc += (sm[p]/den)*v;
  }
  imf[m*64+t]=acc;
}

// ---------------- normalize pc_feats / img_feats ----------------
__global__ void k_norm2(const float* pcf, const float* imf, float* pcn, float* imn){
  __shared__ float red[64];
  int b=blockIdx.x, t=threadIdx.x;
  const float* src; float* dst; int row;
  if (b<NPTS){ src=pcf; dst=pcn; row=b; } else { src=imf; dst=imn; row=b-NPTS; }
  float v=src[row*64+t];
  red[t]=v*v; __syncthreads();
  for (int s=32;s>0;s>>=1){ if(t<s) red[t]+=red[t+s]; __syncthreads(); }
  dst[row*64+t]=v/fmaxf(sqrtf(red[0]),1e-12f);
}

// ---------------- An / Bm (rank-structured layer-1) ----------------
__global__ void k_anbm(const float* xyz, const float* wp, const float* rfit,
                       const float* rf3, const float* w10, const float* b10,
                       float* An, float* Bm){
  int b=blockIdx.x, j=threadIdx.x;
  if (b<NPTS){
    int n=b;
    float a=b10[j];
    for (int c=0;c<3;++c)  a += xyz[n*3+c]*w10[c*128+j];
    for (int c=0;c<64;++c) a += wp[n*64+c]*w10[(6+c)*128+j];
    An[n*128+j]=a;
  } else {
    int m=b-NPTS;
    float a=0.f;
    for (int c=0;c<3;++c)  a += rfit[m*3+c]*w10[(3+c)*128+j];
    for (int c=0;c<64;++c) a += rf3[c*MPIX+m]*w10[(70+c)*128+j];
    Bm[m*128+j]=a;
  }
}

// ---------------- fused big MLP (WMMA) + online softmax + att/corres ----------------
__global__ void __launch_bounds__(128)
k_bigmlp(const float* An, const float* Bm, const float* w10,
         const float* sim1, const float* rmax1, const float* cmax1,
         const float* sim2, const float* rmax2, const float* cmax2,
         const float* w11, const float* b11, const float* w12, const float* b12,
         const float* rfit, float* att, float* cor){
  // weights stored TRANSPOSED [col][k] so each lane's 16 B-fragment halves are
  // contiguous + 32B aligned -> ds_load_b128 pairs instead of 16x ds_load_u16
  __shared__ _Float16 w11s[64*128];   // [col][k]
  __shared__ _Float16 w12s[64*64];    // [col][k]
  __shared__ float Wss[4*128];
  __shared__ float Ans[128];
  __shared__ _Float16 h1s[4][16*128];
  __shared__ _Float16 h2s[4][16*64];
  __shared__ _Float16 h3s[4][16*64];
  __shared__ float pbuf[4][16];
  __shared__ float comb_max[4], comb_d[4];
  __shared__ float comb_att[4][64];
  __shared__ float comb_cor[4][3];

  int n   = blockIdx.x;
  int tid = threadIdx.x, wv = tid>>5, ln = tid&31;

  for (int i=tid;i<64*128;i+=128){ int c=i>>7, k=i&127; w11s[i]=(_Float16)w11[k*64+c]; }
  for (int i=tid;i<64*64; i+=128){ int c=i>>6, k=i&63;  w12s[i]=(_Float16)w12[k*64+c]; }
  for (int i=tid;i<4*128; i+=128)  Wss[i]=w10[134*128+i];   // rows 134..137 of w1_0
  if (tid<128) Ans[tid]=An[n*128+tid];
  __syncthreads();

  float r1 = rmax1[n] + 1e-6f;
  float r2 = rmax2[n] + 1e-6f;
  float run_max=-1e30f, denom=0.f;
  float acc0=0.f, acc1=0.f, accc=0.f;
  int c0 = ln*2;
  int row16  = ln & 15;
  int khalfA = (ln>>4)*16;
  int khalfB = (ln>>4)*16;
  int drow0  = (ln>>4)*8;

  for (int tile = wv; tile < 64; tile += 4) {
    int m0 = tile*16;
    // prefetch this wave's next tile of Bm into cache (global_prefetch_b8)
    if (tile + 4 < 64) {
      int mpf = (tile+4)*16 + row16;
      __builtin_prefetch(&Bm[(size_t)mpf*128 + (size_t)(ln>>4)*64], 0, 1);
    }
    // ---- layer 1 (rank-structured) -> h1s (f16) ----
    for (int r=0;r<16;++r){
      int m = m0+r;
      float sv1 = sim1[(size_t)n*MPIX+m];
      float sv2 = sim2[(size_t)n*MPIX+m];
      float s0 = sv1/r1, s1 = sv1/(cmax1[m]+1e-10f);
      float s2 = sv2/r2, s3 = sv2/(cmax2[m]+1e-10f);
      #pragma unroll
      for (int jj=0;jj<4;++jj){
        int j = ln + jj*32;
        float v = Ans[j] + Bm[(size_t)m*128+j]
                + s0*Wss[j] + s1*Wss[128+j] + s2*Wss[256+j] + s3*Wss[384+j];
        h1s[wv][r*128+j] = (_Float16)fmaxf(v,0.f);
      }
    }
    // ---- layer 2: (16x128)@(128x64) via 16 WMMAs ----
    for (int ct=0;ct<4;++ct){
      int col = ct*16 + row16;
      v8f acc = {0.f,0.f,0.f,0.f,0.f,0.f,0.f,0.f};
      for (int kt=0;kt<4;++kt){
        int kb = kt*32;
        v16h a;
        #pragma unroll
        for (int e=0;e<8;++e){
          a[e]   = h1s[wv][row16*128 + kb + khalfA + e];
          a[e+8] = h1s[wv][row16*128 + kb + khalfA + 8 + e];
        }
        v16h bf = *(const v16h*)(w11s + col*128 + kb + khalfB);
        acc = __builtin_amdgcn_wmma_f32_16x16x32_f16(false, a, false, bf,
                                                     (short)0, acc, false, false);
      }
      float bb = b11[col];
      #pragma unroll
      for (int i=0;i<8;++i)
        h2s[wv][(i+drow0)*64+col] = (_Float16)fmaxf(acc[i]+bb, 0.f);
    }
    // ---- layer 3: (16x64)@(64x64) via 8 WMMAs ----
    for (int ct=0;ct<4;++ct){
      int col = ct*16 + row16;
      v8f acc = {0.f,0.f,0.f,0.f,0.f,0.f,0.f,0.f};
      for (int kt=0;kt<2;++kt){
        int kb = kt*32;
        v16h a;
        #pragma unroll
        for (int e=0;e<8;++e){
          a[e]   = h2s[wv][row16*64 + kb + khalfA + e];
          a[e+8] = h2s[wv][row16*64 + kb + khalfA + 8 + e];
        }
        v16h bf = *(const v16h*)(w12s + col*64 + kb + khalfB);
        acc = __builtin_amdgcn_wmma_f32_16x16x32_f16(false, a, false, bf,
                                                     (short)0, acc, false, false);
      }
      float bb = b12[col];
      #pragma unroll
      for (int i=0;i<8;++i)
        h3s[wv][(i+drow0)*64+col] = (_Float16)fmaxf(acc[i]+bb, 0.f);
    }
    // ---- per-row logits ----
    if (ln < 16){
      float l=-1e30f;
      for (int c=0;c<64;++c) l = fmaxf(l, (float)h3s[wv][ln*64+c]);
      pbuf[wv][ln]=l;
    }
    // ---- online softmax update (lane-uniform) ----
    float lv[16], p[16];
    float tm=-1e30f;
    #pragma unroll
    for (int i=0;i<16;++i){ lv[i]=pbuf[wv][i]; tm=fmaxf(tm,lv[i]); }
    float nm = fmaxf(run_max, tm);
    float sc = __expf(run_max-nm);
    float ps=0.f;
    #pragma unroll
    for (int i=0;i<16;++i){ p[i]=__expf(lv[i]-nm); ps+=p[i]; }
    denom = denom*sc + ps;
    float a0=0.f,a1=0.f;
    #pragma unroll
    for (int i=0;i<16;++i){
      a0 += p[i]*(float)h3s[wv][i*64+c0];
      a1 += p[i]*(float)h3s[wv][i*64+c0+1];
    }
    acc0 = acc0*sc + a0;
    acc1 = acc1*sc + a1;
    if (ln<3){
      float ac=0.f;
      for (int i=0;i<16;++i) ac += p[i]*rfit[(m0+i)*3+ln];
      accc = accc*sc + ac;
    }
    run_max = nm;
  }
  if (ln==0){ comb_max[wv]=run_max; comb_d[wv]=denom; }
  if (ln<3)  comb_cor[wv][ln]=accc;
  comb_att[wv][c0]=acc0; comb_att[wv][c0+1]=acc1;
  __syncthreads();
  if (wv==0){
    float Mg=-1e30f;
    for (int w_=0;w_<4;++w_) Mg=fmaxf(Mg, comb_max[w_]);
    float dt=0.f, f[4];
    for (int w_=0;w_<4;++w_){ f[w_]=__expf(comb_max[w_]-Mg); dt += comb_d[w_]*f[w_]; }
    float o0=0.f,o1=0.f;
    for (int w_=0;w_<4;++w_){ o0 += comb_att[w_][c0]*f[w_]; o1 += comb_att[w_][c0+1]*f[w_]; }
    att[n*64+c0]=o0/dt; att[n*64+c0+1]=o1/dt;
    if (ln<3){
      float oc=0.f;
      for (int w_=0;w_<4;++w_) oc += comb_cor[w_][ln]*f[w_];
      cor[n*3+ln]=oc/dt;
    }
  }
}

// ---------------- head MLP -> weights (d_out+12) and w_in ----------------
__global__ void k_head(const float* att, const float* wm0,const float* bm0,
                       const float* wm1,const float* bm1,const float* wm2,const float* bm2,
                       float* out_w, float* win){
  __shared__ float a0[64], hh[64], h2[128], lg[2];
  int n=blockIdx.x, t=threadIdx.x;
  if (t<64) a0[t]=att[n*64+t];
  __syncthreads();
  if (t<64){
    float a=bm0[t];
    for (int c=0;c<64;++c) a+=a0[c]*wm0[c*64+t];
    hh[t]=fmaxf(a,0.f);
  }
  __syncthreads();
  {
    float a=bm1[t];
    for (int c=0;c<64;++c) a+=hh[c]*wm1[c*128+t];
    h2[t]=fmaxf(a,0.f);
  }
  __syncthreads();
  if (t<2){
    float a=bm2[t];
    for (int c=0;c<128;++c) a+=h2[c]*wm2[c*2+t];
    lg[t]=a;
  }
  __syncthreads();
  if (t==0){
    float mx=fmaxf(lg[0],lg[1]);
    float e0=__expf(lg[0]-mx), e1=__expf(lg[1]-mx);
    float d=e0+e1;
    out_w[n*2]=e0/d; out_w[n*2+1]=e1/d;
    win[n]=(lg[1]>lg[0])?1.f:0.f;
  }
}

// ---------------- weighted Kabsch (3x3 SVD) -> R, t ----------------
__device__ __forceinline__ void blk_reduce_add(float* red, float v, float* dst, int t){
  red[t]=v; __syncthreads();
  for (int s=128;s>0;s>>=1){ if(t<s) red[t]+=red[t+s]; __syncthreads(); }
  if (t==0)*dst=red[0];
  __syncthreads();
}

__global__ void k_pnp(const float* xyz, const float* cor, const float* win, float* out){
  __shared__ float red[256];
  __shared__ float sums[16];
  int t = threadIdx.x;
  float w = win[t];
  float sx=xyz[t*3], sy=xyz[t*3+1], sz=xyz[t*3+2];
  float u=cor[t*3], v=cor[t*3+1];
  float dx=u*sz, dy=v*sz, dz=sz;
  blk_reduce_add(red, w,    &sums[0], t);
  blk_reduce_add(red, w*sx, &sums[1], t);
  blk_reduce_add(red, w*sy, &sums[2], t);
  blk_reduce_add(red, w*sz, &sums[3], t);
  blk_reduce_add(red, w*dx, &sums[4], t);
  blk_reduce_add(red, w*dy, &sums[5], t);
  blk_reduce_add(red, w*dz, &sums[6], t);
  float invw = 1.f/(sums[0]+1e-8f);
  float scx=sums[1]*invw, scy=sums[2]*invw, scz=sums[3]*invw;
  float dcx=sums[4]*invw, dcy=sums[5]*invw, dcz=sums[6]*invw;
  float wn = w*invw;
  float cs[3]={sx-scx, sy-scy, sz-scz};
  float cd[3]={dx-dcx, dy-dcy, dz-dcz};
  for (int i=0;i<3;++i)
    for (int j=0;j<3;++j)
      blk_reduce_add(red, wn*cs[i]*cd[j], &sums[7+i*3+j], t);
  if (t==0){
    float H[3][3], A[3][3], V[3][3]={{1,0,0},{0,1,0},{0,0,1}};
    for (int i=0;i<3;++i) for (int j=0;j<3;++j) H[i][j]=sums[7+i*3+j];
    for (int i=0;i<3;++i) for (int j=0;j<3;++j)
      A[i][j]=H[0][i]*H[0][j]+H[1][i]*H[1][j]+H[2][i]*H[2][j];
    const int pqs[3][2]={{0,1},{0,2},{1,2}};
    for (int sweep=0;sweep<16;++sweep){
      for (int r=0;r<3;++r){
        int p=pqs[r][0], q=pqs[r][1];
        float apq=A[p][q];
        if (fabsf(apq) < 1e-14f) continue;
        float th=(A[q][q]-A[p][p])/(2.f*apq);
        float tt=((th>=0.f)?1.f:-1.f)/(fabsf(th)+sqrtf(th*th+1.f));
        float c=1.f/sqrtf(tt*tt+1.f), s=tt*c;
        for (int k=0;k<3;++k){ float a1=A[k][p],a2=A[k][q]; A[k][p]=c*a1-s*a2; A[k][q]=s*a1+c*a2; }
        for (int k=0;k<3;++k){ float a1=A[p][k],a2=A[q][k]; A[p][k]=c*a1-s*a2; A[q][k]=s*a1+c*a2; }
        for (int k=0;k<3;++k){ float v1=V[k][p],v2=V[k][q]; V[k][p]=c*v1-s*v2; V[k][q]=s*v1+c*v2; }
      }
    }
    float lam[3]={A[0][0],A[1][1],A[2][2]};
    int ord[3]={0,1,2}; int tmp;
    if (lam[ord[0]]<lam[ord[1]]){tmp=ord[0];ord[0]=ord[1];ord[1]=tmp;}
    if (lam[ord[0]]<lam[ord[2]]){tmp=ord[0];ord[0]=ord[2];ord[2]=tmp;}
    if (lam[ord[1]]<lam[ord[2]]){tmp=ord[1];ord[1]=ord[2];ord[2]=tmp;}
    float Vv[3][3], sig[3], U[3][3];
    for (int k=0;k<3;++k){
      sig[k]=sqrtf(fmaxf(lam[ord[k]],0.f));
      for (int i=0;i<3;++i) Vv[i][k]=V[i][ord[k]];
    }
    for (int k=0;k<3;++k){
      float u0=H[0][0]*Vv[0][k]+H[0][1]*Vv[1][k]+H[0][2]*Vv[2][k];
      float u1=H[1][0]*Vv[0][k]+H[1][1]*Vv[1][k]+H[1][2]*Vv[2][k];
      float u2=H[2][0]*Vv[0][k]+H[2][1]*Vv[1][k]+H[2][2]*Vv[2][k];
      if (sig[k]>1e-10f){ U[0][k]=u0/sig[k]; U[1][k]=u1/sig[k]; U[2][k]=u2/sig[k]; }
      else { U[0][k]=0.f; U[1][k]=0.f; U[2][k]=0.f; }
    }
    if (sig[0]<=1e-10f){ U[0][0]=1.f; U[1][0]=0.f; U[2][0]=0.f; }
    if (sig[1]<=1e-10f){
      float a0=U[0][0],a1=U[1][0],a2=U[2][0], c0,c1,c2;
      if (fabsf(a0)<0.9f){ c0=0.f; c1=a2; c2=-a1; } else { c0=-a2; c1=0.f; c2=a0; }
      float nn=sqrtf(c0*c0+c1*c1+c2*c2)+1e-20f;
      U[0][1]=c0/nn; U[1][1]=c1/nn; U[2][1]=c2/nn;
    }
    if (sig[2]<=1e-10f){
      float c0=U[1][0]*U[2][1]-U[2][0]*U[1][1];
      float c1=U[2][0]*U[0][1]-U[0][0]*U[2][1];
      float c2=U[0][0]*U[1][1]-U[1][0]*U[0][1];
      float nn=sqrtf(c0*c0+c1*c1+c2*c2)+1e-20f;
      U[0][2]=c0/nn; U[1][2]=c1/nn; U[2][2]=c2/nn;
    }
    float Mm[3][3];
    for (int i=0;i<3;++i) for (int j=0;j<3;++j)
      Mm[i][j]=Vv[i][0]*U[j][0]+Vv[i][1]*U[j][1]+Vv[i][2]*U[j][2];
    float det = Mm[0][0]*(Mm[1][1]*Mm[2][2]-Mm[1][2]*Mm[2][1])
              - Mm[0][1]*(Mm[1][0]*Mm[2][2]-Mm[1][2]*Mm[2][0])
              + Mm[0][2]*(Mm[1][0]*Mm[2][1]-Mm[1][1]*Mm[2][0]);
    float R[3][3];
    for (int i=0;i<3;++i) for (int j=0;j<3;++j)
      R[i][j]=Vv[i][0]*U[j][0]+Vv[i][1]*U[j][1]+det*Vv[i][2]*U[j][2];
    for (int i=0;i<3;++i) for (int j=0;j<3;++j) out[i*3+j]=R[i][j];
    float dc[3]={dcx,dcy,dcz}, sc3[3]={scx,scy,scz};
    for (int i=0;i<3;++i)
      out[9+i]=dc[i]-(R[i][0]*sc3[0]+R[i][1]*sc3[1]+R[i][2]*sc3[2]);
  }
}

extern "C" void kernel_launch(void* const* d_in, const int* in_sizes, int n_in,
                              void* d_out, int out_size, void* d_ws, size_t ws_size,
                              hipStream_t stream) {
  (void)in_sizes; (void)n_in; (void)out_size; (void)ws_size;
  const float* wxyz=(const float*)d_in[0];
  const float* wp  =(const float*)d_in[1];
  const float* rf3 =(const float*)d_in[2];
  const float* rf3i=(const float*)d_in[3];
  const float* lz  =(const float*)d_in[4];
  const float* w20=(const float*)d_in[5],  *b20=(const float*)d_in[6];
  const float* w21=(const float*)d_in[7],  *b21=(const float*)d_in[8];
  const float* w22=(const float*)d_in[9],  *b22=(const float*)d_in[10];
  const float* w30=(const float*)d_in[11], *b30=(const float*)d_in[12];
  const float* w31=(const float*)d_in[13], *b31=(const float*)d_in[14];
  const float* w32=(const float*)d_in[15], *b32=(const float*)d_in[16];
  const float* w10=(const float*)d_in[17], *b10=(const float*)d_in[18];
  const float* w11=(const float*)d_in[19], *b11=(const float*)d_in[20];
  const float* w12=(const float*)d_in[21], *b12=(const float*)d_in[22];
  const float* wm0=(const float*)d_in[23], *bm0=(const float*)d_in[24];
  const float* wm1=(const float*)d_in[25], *bm1=(const float*)d_in[26];
  const float* wm2=(const float*)d_in[27], *bm2=(const float*)d_in[28];
  float* out=(float*)d_out;
  float* ws =(float*)d_ws;

  float* xyz =ws+OXYZ; float* snorm=ws+OSN;  float* dnorm=ws+ODN;  float* rfit=ws+ORFI;
  float* sim1=ws+OS1;  float* rm1=ws+ORM1;   float* cm1=ws+OCM1;
  float* pcf =ws+OPCF; float* pcn=ws+OPCN;   float* imf=ws+OIMF;   float* imn=ws+OIMN;
  float* sim2=ws+OS2;  float* rm2=ws+ORM2;   float* cm2=ws+OCM2;
  float* An  =ws+OAN;  float* Bm =ws+OBM;
  float* att =ws+OATT; float* cor=ws+OCOR;   float* win=ws+OWIN;

  k_prep <<<dim3(1280),dim3(64), 0,stream>>>(wxyz,wp,rf3,rf3i,lz,xyz,snorm,dnorm,rfit);
  k_sim  <<<dim3(256), dim3(256),0,stream>>>(snorm,dnorm,sim1);
  k_maxes<<<dim3(1280),dim3(256),0,stream>>>(sim1,rm1,cm1);
  k_knn  <<<dim3(256), dim3(64), 0,stream>>>(xyz,wp,w20,b20,w21,b21,w22,b22,pcf);
  k_img  <<<dim3(1024),dim3(64), 0,stream>>>(rf3,rf3i,w30,b30,w31,b31,w32,b32,imf);
  k_norm2<<<dim3(1280),dim3(64), 0,stream>>>(pcf,imf,pcn,imn);
  k_sim  <<<dim3(256), dim3(256),0,stream>>>(pcn,imn,sim2);
  k_maxes<<<dim3(1280),dim3(256),0,stream>>>(sim2,rm2,cm2);
  k_anbm <<<dim3(1280),dim3(128),0,stream>>>(xyz,wp,rfit,rf3,w10,b10,An,Bm);
  k_bigmlp<<<dim3(256),dim3(128),0,stream>>>(An,Bm,w10,sim1,rm1,cm1,sim2,rm2,cm2,
                                             w11,b11,w12,b12,rfit,att,cor);
  k_head <<<dim3(256), dim3(128),0,stream>>>(att,wm0,bm0,wm1,bm1,wm2,bm2,out+12,win);
  k_pnp  <<<dim3(1),   dim3(256),0,stream>>>(xyz,cor,win,out);
}
